// HardSOM_927712936091
// MI455X (gfx1250) — matrix operand order repaired and checked by reading, not compiled, not run on hardware
//
#include <hip/hip_runtime.h>
#include <math.h>

// ---------------------------------------------------------------------------
// VQ (HardSOM) for MI455X / gfx1250: bf16x3 WMMA distance argmin.
// N=65536 rows, D=512, K=2048 codes. Compute-bound: 1.37e11 FLOP vs 0.27 GB.
// ---------------------------------------------------------------------------

#define DDIM   512
#define KCODES 2048
#define NROWS  65536
#define ROWS_PER_BLOCK 128   // 8 waves x 16 rows

typedef __attribute__((ext_vector_type(16))) __bf16 v16bf;
typedef __attribute__((ext_vector_type(8)))  __bf16 v8bf;
typedef __attribute__((ext_vector_type(8)))  float  v8f;

struct alignas(8) us4 { unsigned short x, y, z, w; };

__device__ __forceinline__ unsigned short f32_bf16_rne(float f) {
    unsigned int u = __float_as_uint(f);
    u += 0x7fffu + ((u >> 16) & 1u);          // round-to-nearest-even
    return (unsigned short)(u >> 16);
}
__device__ __forceinline__ float bf16_f32(unsigned short h) {
    return __uint_as_float(((unsigned int)h) << 16);
}

// --------------------------- init scratch ----------------------------------
__global__ void vq_init_kernel(unsigned int* __restrict__ hist,
                               double* __restrict__ lossAcc) {
    int t = blockIdx.x * blockDim.x + threadIdx.x;
    if (t < KCODES) hist[t] = 0u;
    if (t == 0)     *lossAcc = 0.0;
}

// ------------------- codebook: bf16 hi/lo split + ||w||^2 ------------------
__global__ void __launch_bounds__(128) vq_prep_codebook(
    const float* __restrict__ w,
    unsigned short* __restrict__ wh, unsigned short* __restrict__ wl,
    float* __restrict__ wn2)
{
    const int row = blockIdx.x;          // K rows
    const int t   = threadIdx.x;         // 128 threads x 4 cols
    const float4 v = ((const float4*)(w + (size_t)row * DDIM))[t];
    float f[4] = { v.x, v.y, v.z, v.w };
    us4 hh, ll;
    unsigned short* hp = &hh.x;
    unsigned short* lp = &ll.x;
    float s = 0.f;
#pragma unroll
    for (int j = 0; j < 4; ++j) {
        s += f[j] * f[j];
        unsigned short h = f32_bf16_rne(f[j]);
        hp[j] = h;
        lp[j] = f32_bf16_rne(f[j] - bf16_f32(h));
    }
    *(us4*)(wh + (size_t)row * DDIM + t * 4) = hh;
    *(us4*)(wl + (size_t)row * DDIM + t * 4) = ll;
#pragma unroll
    for (int off = 16; off >= 1; off >>= 1) s += __shfl_xor(s, off, 32);
    __shared__ float sh[4];
    if ((t & 31) == 0) sh[t >> 5] = s;
    __syncthreads();
    if (t == 0) wn2[row] = sh[0] + sh[1] + sh[2] + sh[3];
}

// ------------------- main: WMMA distance + per-row argmin ------------------
// Block = 256 threads (8 waves). Each wave owns 16 rows; block stages 128 rows
// of x as bf16 hi/lo in dynamic LDS (2 * 128*512*2B = 256 KB <= 320 KB WGP).
__global__ void __launch_bounds__(256) vq_argmin_kernel(
    const float* __restrict__ x,
    const unsigned short* __restrict__ wh,
    const unsigned short* __restrict__ wl,
    const float* __restrict__ wn2,
    int* __restrict__ out_idx)
{
    extern __shared__ unsigned short smem[];
    unsigned short* Ah = smem;                         // [128][512] hi
    unsigned short* Al = smem + ROWS_PER_BLOCK * DDIM; // [128][512] lo

    const int tid     = threadIdx.x;
    const int rowBase = blockIdx.x * ROWS_PER_BLOCK;

    // ---- stage + split x tile (coalesced float4 loads) ----
    for (int i = tid; i < (ROWS_PER_BLOCK * DDIM) / 4; i += 256) {
        const float4 v = ((const float4*)(x + (size_t)rowBase * DDIM))[i];
        float f[4] = { v.x, v.y, v.z, v.w };
        us4 hh, ll;
        unsigned short* hp = &hh.x;
        unsigned short* lp = &ll.x;
#pragma unroll
        for (int j = 0; j < 4; ++j) {
            unsigned short h = f32_bf16_rne(f[j]);
            hp[j] = h;
            lp[j] = f32_bf16_rne(f[j] - bf16_f32(h));
        }
        *(us4*)(Ah + i * 4) = hh;
        *(us4*)(Al + i * 4) = ll;
    }
    __syncthreads();

    const int  wave  = tid >> 5;
    const int  lane  = tid & 31;
    const int  lhalf = lane & 15;
    const bool upper = lane >= 16;
    const int  aRow  = wave * 16 + lhalf;   // A-matrix row M = lane&15 (ISA 7.12.2)
    const int  uoffA = upper ? 16 : 0;      // upper lanes: K+16 chunks
    const int  uoffB = upper ? 16 : 0;      // B: upper lanes hold K+16 rows

    float best[8];
    int   bidx[8];
#pragma unroll
    for (int r = 0; r < 8; ++r) { best[r] = 3.4e38f; bidx[r] = 0; }

    for (int nt = 0; nt < KCODES / 16; ++nt) {
        const int   n   = nt * 16 + lhalf;   // C/D column owned by this lane
        const float wnv = wn2[n];
        const unsigned short* bhp = wh + (size_t)n * DDIM + uoffB;
        const unsigned short* blp = wl + (size_t)n * DDIM + uoffB;

        v8f c = { 0.f, 0.f, 0.f, 0.f, 0.f, 0.f, 0.f, 0.f };
#pragma unroll 4
        for (int s = 0; s < DDIM / 32; ++s) {   // 16 K-steps of 32
            const int kb = s * 32;
            // A fragment: elements 0..7 -> K kb..kb+7(+16), 8..15 -> kb+8..15(+16)
            v8bf a0h = *(const v8bf*)(Ah + aRow * DDIM + kb + uoffA);
            v8bf a1h = *(const v8bf*)(Ah + aRow * DDIM + kb + 8 + uoffA);
            v8bf a0l = *(const v8bf*)(Al + aRow * DDIM + kb + uoffA);
            v8bf a1l = *(const v8bf*)(Al + aRow * DDIM + kb + 8 + uoffA);
            v16bf ah, al;
#pragma unroll
            for (int e = 0; e < 8; ++e) {
                ah[e] = a0h[e]; ah[e + 8] = a1h[e];
                al[e] = a0l[e]; al[e + 8] = a1l[e];
            }
            // B fragment: half-slot e = row K kb+e (+16 upper lanes), col n
            v16bf bh = *(const v16bf*)(bhp + kb);
            v16bf bl = *(const v16bf*)(blp + kb);
            // bf16x3 emulation of f32 dot: hi*hi + hi*lo + lo*hi
            c = __builtin_amdgcn_wmma_f32_16x16x32_bf16(false, ah, false, bh,
                                                        (short)0, c, false, false);
            c = __builtin_amdgcn_wmma_f32_16x16x32_bf16(false, ah, false, bl,
                                                        (short)0, c, false, false);
            c = __builtin_amdgcn_wmma_f32_16x16x32_bf16(false, al, false, bh,
                                                        (short)0, c, false, false);
        }
        // distance (minus constant ||x||^2): ||w||^2 - 2 x.w  — exact wn2, bf16x3 dot
#pragma unroll
        for (int r = 0; r < 8; ++r) {
            const float dist = wnv - 2.0f * c[r];
            if (dist < best[r]) { best[r] = dist; bidx[r] = n; }
        }
    }

    // argmin across the 16 lanes that share a row (xor offsets < 16 keep halves)
#pragma unroll
    for (int off = 8; off >= 1; off >>= 1) {
#pragma unroll
        for (int r = 0; r < 8; ++r) {
            const float ob = __shfl_xor(best[r], off, 32);
            const int   oi = __shfl_xor(bidx[r], off, 32);
            if (ob < best[r] || (ob == best[r] && oi < bidx[r])) {
                best[r] = ob; bidx[r] = oi;
            }
        }
    }
    if (lhalf == 0) {
        const int rBase = rowBase + wave * 16 + (upper ? 8 : 0);
#pragma unroll
        for (int r = 0; r < 8; ++r) out_idx[rBase + r] = bidx[r];
    }
}

// ------------- gather + exact f32 loss + histogram + idx output ------------
__global__ void __launch_bounds__(128) vq_gather_kernel(
    const float* __restrict__ x, const float* __restrict__ w,
    const int* __restrict__ idx,
    float* __restrict__ outq, float* __restrict__ outIdxF,
    unsigned int* __restrict__ hist, double* __restrict__ lossAcc)
{
    const int row = blockIdx.x;
    const int t   = threadIdx.x;      // 128 threads x float4
    const int k   = idx[row];
    const float4 wv = ((const float4*)(w + (size_t)k   * DDIM))[t];
    const float4 xv = ((const float4*)(x + (size_t)row * DDIM))[t];
    ((float4*)(outq + (size_t)row * DDIM))[t] = wv;   // quantized_st == quantized
    const float d0 = wv.x - xv.x, d1 = wv.y - xv.y;
    const float d2 = wv.z - xv.z, d3 = wv.w - xv.w;
    float s = d0 * d0 + d1 * d1 + d2 * d2 + d3 * d3;
#pragma unroll
    for (int off = 16; off >= 1; off >>= 1) s += __shfl_xor(s, off, 32);
    __shared__ float sh[4];
    if ((t & 31) == 0) sh[t >> 5] = s;
    __syncthreads();
    if (t == 0) {
        atomicAdd(lossAcc, (double)(sh[0] + sh[1] + sh[2] + sh[3]));
        atomicAdd(&hist[k], 1u);
        outIdxF[row] = (float)k;
    }
}

// --------------------------- perplexity + loss -----------------------------
__global__ void __launch_bounds__(256) vq_finalize_kernel(
    const unsigned int* __restrict__ hist, const double* __restrict__ lossAcc,
    float* __restrict__ outLoss, float* __restrict__ outPerp)
{
    const int t = threadIdx.x;
    float h = 0.f;
    for (int i = t; i < KCODES; i += 256) {
        const float p = (float)hist[i] * (1.0f / (float)NROWS);
        h -= p * logf(p + 1e-10f);
    }
#pragma unroll
    for (int off = 16; off >= 1; off >>= 1) h += __shfl_xor(h, off, 32);
    __shared__ float sh[8];
    if ((t & 31) == 0) sh[t >> 5] = h;
    __syncthreads();
    if (t == 0) {
        float H = 0.f;
#pragma unroll
        for (int i = 0; i < 8; ++i) H += sh[i];
        *outPerp = expf(H);
        *outLoss = (float)(*lossAcc / ((double)NROWS * (double)DDIM)) * 0.25f;
    }
}

// ---------------------------------------------------------------------------
extern "C" void kernel_launch(void* const* d_in, const int* in_sizes, int n_in,
                              void* d_out, int out_size, void* d_ws, size_t ws_size,
                              hipStream_t stream) {
    (void)in_sizes; (void)n_in; (void)out_size; (void)ws_size;
    const float* x = (const float*)d_in[0];   // [B,T,D] f32
    const float* w = (const float*)d_in[1];   // [K,D]   f32
    float* out = (float*)d_out;

    // workspace layout
    char* ws = (char*)d_ws;
    unsigned short* wh   = (unsigned short*)(ws);                 // 2 MB
    unsigned short* wl   = (unsigned short*)(ws + 2097152);       // 2 MB
    float*          wn2  = (float*)        (ws + 4194304);        // 8 KB
    int*            idx  = (int*)          (ws + 4202496);        // 256 KB
    unsigned int*   hist = (unsigned int*) (ws + 4464640);        // 8 KB
    double*       lossAc = (double*)       (ws + 4472832);        // 8 B

    // output layout: loss | quantized_st (N*D) | perplexity | indices (N)
    float* outLoss = out;
    float* outQ    = out + 1;
    float* outPerp = out + 1 + (size_t)NROWS * DDIM;
    float* outIdx  = out + 2 + (size_t)NROWS * DDIM;

    vq_init_kernel<<<(KCODES + 255) / 256, 256, 0, stream>>>(hist, lossAc);
    vq_prep_codebook<<<KCODES, 128, 0, stream>>>(w, wh, wl, wn2);

    const size_t ldsBytes = 2u * ROWS_PER_BLOCK * DDIM * sizeof(unsigned short); // 256 KB
    vq_argmin_kernel<<<NROWS / ROWS_PER_BLOCK, 256, ldsBytes, stream>>>(
        x, wh, wl, wn2, idx);

    vq_gather_kernel<<<NROWS, 128, 0, stream>>>(x, w, idx, outQ, outIdx, hist, lossAc);
    vq_finalize_kernel<<<1, 256, 0, stream>>>(hist, lossAc, outLoss, outPerp);
}
